// Encoder_5677946765351
// MI455X (gfx1250) — matrix operand order repaired and checked by reading, not compiled, not run on hardware
//
#include <hip/hip_runtime.h>
#include <hip/hip_bf16.h>
#include <math.h>

// ---------------- model constants ----------------
#define LAYERS 12
#define HEADS  12
#define DMODEL 768
#define HDIM   64
#define DFF    3072
#define NTOK   512          // tokens per batch item
#define BATCH  4
#define BN     (BATCH*NTOK) // 2048 rows
#define KPATCH 16384        // C*P^3
#define NPIX   33554432L    // BN*KPATCH

// ---------------- WMMA types ----------------
typedef __attribute__((ext_vector_type(16))) __bf16 v16bf;
typedef __attribute__((ext_vector_type(8)))  float  v8f;

union FragBF { v16bf v; uint4 q[2]; };

__device__ __forceinline__ unsigned short f2bf(float f) {
  unsigned int u = __float_as_uint(f);
  unsigned int r = u + 0x7FFFu + ((u >> 16) & 1u);  // round-to-nearest-even
  return (unsigned short)(r >> 16);
}

// ---------------- generic batched bf16 WMMA GEMM ----------------
// C[M,N] (f32 acc) = alpha * A[M,K](bf16,row-major,lda) * B
//   BMODE==0 : B given as [N][K] bf16 (pre-transposed weights / K-matrix), ldb = row stride
//   BMODE==1 : B given as [K][N] bf16 (V matrix), ldb = row stride
// EMODE: 0 = f32 store (+bias)   1 = f32 residual add (+bias)
//        2 = bf16 store (+bias)  3 = bf16 gelu(acc+bias)
#define TILE_M 64
#define TILE_N 64
#define TILE_K 64   // two WMMA K-substeps per staging round
#define LSTR   72   // padded LDS row stride in bf16 units (144B = 9*16B)

template <int BMODE, int EMODE>
__global__ __launch_bounds__(128, 1) void gemm_bf16_wmma(
    const unsigned short* __restrict__ A,
    const unsigned short* __restrict__ B,
    float* __restrict__ fOut,
    unsigned short* __restrict__ bfOut,
    const float* __restrict__ bias,
    int M, int N, int K,
    int lda, int ldb, int ldc,
    float alpha,
    int zdiv,
    long aZ1, long aZ2, long bZ1, long bZ2, long cZ1, long cZ2)
{
  // double-buffered tiles: one barrier per K-step
  __shared__ alignas(16) unsigned short As[2][TILE_M * LSTR];
  __shared__ alignas(16) unsigned short Bs[2][TILE_N * LSTR];

  const int z  = blockIdx.z;
  const long zq = z / zdiv, zr = z - zq * zdiv;
  A += zq * aZ1 + zr * aZ2;
  B += zq * bZ1 + zr * bZ2;
  const long cOff = zq * cZ1 + zr * cZ2;

  const int tileM = blockIdx.y * TILE_M;
  const int tileN = blockIdx.x * TILE_N;

  const int t    = threadIdx.x;
  const int lane = t & 31;
  const int wave = t >> 5;
  const int wR   = (wave >> 1) * 32;   // wave row offset inside tile
  const int wC   = (wave & 1)  * 32;   // wave col offset inside tile

  v8f acc[2][2] = {};

  // staging thread map: 64 rows x 64 bf16 per tile, 32 bf16 (4x uint4) per thread
  const int sRow = t >> 1;           // 0..63
  const int sCol = (t & 1) * 32;     // bf16 col offset 0/32

  const unsigned short* aPtr = A + (long)(tileM + sRow) * lda + sCol;
  // BMODE 0: row of Bt;  BMODE 1: row k of B (scattered on LDS store)
  const unsigned short* bPtr = (BMODE == 0)
      ? (B + (long)(tileN + sRow) * ldb + sCol)
      : (B + (long)sRow * ldb + tileN + sCol);

  // ---- preload tile 0 into buffer 0 ----
  {
    const uint4* sa = (const uint4*)aPtr;
    const uint4* sb = (const uint4*)bPtr;
    uint4 a0 = sa[0], a1 = sa[1], a2 = sa[2], a3 = sa[3];
    uint4 b0 = sb[0], b1 = sb[1], b2 = sb[2], b3 = sb[3];
    uint4* dA = (uint4*)(&As[0][sRow * LSTR + sCol]);
    dA[0] = a0; dA[1] = a1; dA[2] = a2; dA[3] = a3;
    if (BMODE == 0) {
      uint4* dB = (uint4*)(&Bs[0][sRow * LSTR + sCol]);
      dB[0] = b0; dB[1] = b1; dB[2] = b2; dB[3] = b3;
    } else {
      union { uint4 q[4]; unsigned short s[32]; } tmp;
      tmp.q[0] = b0; tmp.q[1] = b1; tmp.q[2] = b2; tmp.q[3] = b3;
      #pragma unroll
      for (int i = 0; i < 32; ++i) Bs[0][(sCol + i) * LSTR + sRow] = tmp.s[i];
    }
  }
  __syncthreads();

  const int m16 = lane & 15;
  const int kbA = (lane >> 4) * 8;    // A: K-half select within 32-wide substep
  const int kbB = (lane >> 4) * 8;    // B: K-half select within 32-wide substep

  int cur = 0;
  for (int kk = 0; kk < K; kk += TILE_K) {
    // ---- issue next-tile global loads unconditionally (clamped addr on last iter) ----
    const int kn = (kk + TILE_K < K) ? (kk + TILE_K) : kk;
    const uint4* sa = (const uint4*)(aPtr + kn);
    const uint4* sb = (BMODE == 0) ? (const uint4*)(bPtr + kn)
                                   : (const uint4*)(bPtr + (long)kn * ldb);
    uint4 na0 = sa[0], na1 = sa[1], na2 = sa[2], na3 = sa[3];
    uint4 nb0 = sb[0], nb1 = sb[1], nb2 = sb[2], nb3 = sb[3];

    // ---- 2 K-substeps x 4 WMMAs from current buffer (ISA 7.12.2 layouts) ----
    const unsigned short* Ac = &As[cur][0];
    const unsigned short* Bc = &Bs[cur][0];
    #pragma unroll
    for (int ks = 0; ks < 2; ++ks) {
      const int kOff = ks * 32;
      FragBF a[2], b[2];
      #pragma unroll
      for (int i = 0; i < 2; ++i) {
        const unsigned short* base = Ac + (wR + i * 16 + m16) * LSTR + kOff;
        a[i].q[0] = *(const uint4*)(base + kbA);        // e0..7  -> K = kOff+kbA..
        a[i].q[1] = *(const uint4*)(base + 16 + kbA);   // e8..15 -> K = kOff+16+kbA..
      }
      #pragma unroll
      for (int j = 0; j < 2; ++j) {
        const unsigned short* base = Bc + (wC + j * 16 + m16) * LSTR + kOff + kbB;
        b[j].q[0] = *(const uint4*)(base);
        b[j].q[1] = *(const uint4*)(base + 16);
      }
      #pragma unroll
      for (int i = 0; i < 2; ++i)
        #pragma unroll
        for (int j = 0; j < 2; ++j)
          acc[i][j] = __builtin_amdgcn_wmma_f32_16x16x32_bf16(
              false, a[i].v, false, b[j].v, (short)0, acc[i][j], false, false);
    }

    // ---- commit next tile into the other buffer ----
    const int nxt = cur ^ 1;
    {
      uint4* dA = (uint4*)(&As[nxt][sRow * LSTR + sCol]);
      dA[0] = na0; dA[1] = na1; dA[2] = na2; dA[3] = na3;
    }
    if (BMODE == 0) {
      uint4* dB = (uint4*)(&Bs[nxt][sRow * LSTR + sCol]);
      dB[0] = nb0; dB[1] = nb1; dB[2] = nb2; dB[3] = nb3;
    } else {
      union { uint4 q[4]; unsigned short s[32]; } tmp;
      tmp.q[0] = nb0; tmp.q[1] = nb1; tmp.q[2] = nb2; tmp.q[3] = nb3;
      #pragma unroll
      for (int i = 0; i < 32; ++i) Bs[nxt][(sCol + i) * LSTR + sRow] = tmp.s[i];
    }
    __syncthreads();
    cur = nxt;
  }

  // ---- epilogue: C layout m=(lane>>4)*8+r, n=lane&15 ----
  const int n16   = lane & 15;
  const int mBase = (lane >> 4) * 8;
  #pragma unroll
  for (int i = 0; i < 2; ++i) {
    #pragma unroll
    for (int j = 0; j < 2; ++j) {
      const int gn = tileN + wC + j * 16 + n16;
      const float bv = bias ? bias[gn] : 0.0f;
      #pragma unroll
      for (int r = 0; r < 8; ++r) {
        const int gm = tileM + wR + i * 16 + mBase + r;
        float val = acc[i][j][r] * alpha + bv;
        const long off = cOff + (long)gm * ldc + gn;
        if (EMODE == 0) {
          fOut[off] = val;
        } else if (EMODE == 1) {
          fOut[off] += val;
        } else if (EMODE == 2) {
          bfOut[off] = f2bf(val);
        } else {
          float x3 = val * val * val;
          float g  = 0.5f * val * (1.0f + tanhf(0.7978845608f * (val + 0.044715f * x3)));
          bfOut[off] = f2bf(g);
        }
      }
    }
  }
}

// ---------------- im2col (patch gather) -> bf16 A matrix [2048][16384] ----------------
__global__ void im2col_bf16(const float* __restrict__ x, unsigned short* __restrict__ Ap) {
  long idx = (long)blockIdx.x * 256 + threadIdx.x;
  if (idx >= NPIX) return;
  int kidx = (int)(idx & 16383);
  int tok  = (int)(idx >> 14);          // b*512 + n
  int b = tok >> 9, n = tok & 511;
  int d = n >> 6, h = (n >> 3) & 7, w = n & 7;
  int c = kidx >> 12, i = (kidx >> 8) & 15, j = (kidx >> 4) & 15, k = kidx & 15;
  long src = (((long)(b * 4 + c) * 128 + (d * 16 + i)) * 128 + (h * 16 + j)) * 128 + (w * 16 + k);
  Ap[idx] = f2bf(x[src]);
}

// ---------------- weight conversion ----------------
__global__ void flat_cvt_bf16(const float* __restrict__ in, unsigned short* __restrict__ out, long total) {
  long idx = (long)blockIdx.x * 256 + threadIdx.x;
  if (idx < total) out[idx] = f2bf(in[idx]);
}

// in [batch][R][Cc] f32 -> out [batch][Cc][R] bf16
__global__ void transpose_cvt_bf16(const float* __restrict__ in, unsigned short* __restrict__ out,
                                   int R, int Cc, long total) {
  long idx = (long)blockIdx.x * 256 + threadIdx.x;
  if (idx >= total) return;
  long per = (long)R * Cc;
  long bI  = idx / per;
  long rem = idx - bI * per;
  int r = (int)(rem / Cc);
  int c = (int)(rem - (long)r * Cc);
  out[bI * per + (long)c * R + r] = f2bf(in[idx]);
}

// ---------------- LayerNorm: f32 row -> bf16 row ----------------
__global__ __launch_bounds__(256) void layernorm_bf16(
    const float* __restrict__ h, const float* __restrict__ g,
    const float* __restrict__ bta, unsigned short* __restrict__ y) {
  __shared__ float red[256];
  const int row = blockIdx.x, t = threadIdx.x;
  const float* hr = h + (long)row * DMODEL;
  float s = 0.f;
  for (int i = t; i < DMODEL; i += 256) s += hr[i];
  red[t] = s; __syncthreads();
  for (int o = 128; o > 0; o >>= 1) { if (t < o) red[t] += red[t + o]; __syncthreads(); }
  const float mu = red[0] * (1.0f / DMODEL);
  __syncthreads();
  float v = 0.f;
  for (int i = t; i < DMODEL; i += 256) { float d = hr[i] - mu; v += d * d; }
  red[t] = v; __syncthreads();
  for (int o = 128; o > 0; o >>= 1) { if (t < o) red[t] += red[t + o]; __syncthreads(); }
  const float rs = rsqrtf(red[0] * (1.0f / DMODEL) + 1e-5f);
  for (int i = t; i < DMODEL; i += 256)
    y[(long)row * DMODEL + i] = f2bf((hr[i] - mu) * rs * g[i] + bta[i]);
}

// ---------------- row softmax: f32 scores -> bf16 probs ----------------
__global__ __launch_bounds__(128) void softmax_bf16(
    const float* __restrict__ S, unsigned short* __restrict__ P) {
  __shared__ float red[128];
  __shared__ float buf[NTOK];
  const long row = blockIdx.x;
  const float* s = S + row * NTOK;
  const int t = threadIdx.x;
  float m = -3.0e38f;
  for (int i = t; i < NTOK; i += 128) { float v = s[i]; buf[i] = v; m = fmaxf(m, v); }
  red[t] = m; __syncthreads();
  for (int o = 64; o > 0; o >>= 1) { if (t < o) red[t] = fmaxf(red[t], red[t + o]); __syncthreads(); }
  m = red[0]; __syncthreads();
  float sum = 0.f;
  for (int i = t; i < NTOK; i += 128) { float e = expf(buf[i] - m); buf[i] = e; sum += e; }
  red[t] = sum; __syncthreads();
  for (int o = 64; o > 0; o >>= 1) { if (t < o) red[t] += red[t + o]; __syncthreads(); }
  const float inv = 1.0f / red[0];
  for (int i = t; i < NTOK; i += 128) P[row * NTOK + i] = f2bf(buf[i] * inv);
}

// ---------------- misc elementwise ----------------
__global__ void add_pos_kernel(float* __restrict__ h, const float* __restrict__ pos) {
  long idx = (long)blockIdx.x * 256 + threadIdx.x;
  if (idx >= (long)BN * DMODEL) return;
  h[idx] += pos[idx % ((long)NTOK * DMODEL)];
}

__global__ void copy_f32(const float* __restrict__ src, float* __restrict__ dst, long total) {
  long idx = (long)blockIdx.x * 256 + threadIdx.x;
  if (idx < total) dst[idx] = src[idx];
}

// ---------------- host orchestration ----------------
template <int BMODE, int EMODE>
static inline void launch_gemm_t(hipStream_t stream,
    const unsigned short* A, const unsigned short* B,
    float* fOut, unsigned short* bfOut, const float* bias,
    int M, int N, int K, int lda, int ldb, int ldc,
    float alpha, int batches, int zdiv,
    long aZ1, long aZ2, long bZ1, long bZ2, long cZ1, long cZ2) {
  dim3 grid(N / TILE_N, M / TILE_M, batches);
  gemm_bf16_wmma<BMODE, EMODE><<<grid, 128, 0, stream>>>(A, B, fOut, bfOut, bias,
      M, N, K, lda, ldb, ldc, alpha, zdiv, aZ1, aZ2, bZ1, bZ2, cZ1, cZ2);
}

extern "C" void kernel_launch(void* const* d_in, const int* in_sizes, int n_in,
                              void* d_out, int out_size, void* d_ws, size_t ws_size,
                              hipStream_t stream) {
  const float* x     = (const float*)d_in[0];
  const float* Wp    = (const float*)d_in[1];
  const float* bp    = (const float*)d_in[2];
  const float* pos   = (const float*)d_in[3];
  const float* ln1_g = (const float*)d_in[4];
  const float* ln1_b = (const float*)d_in[5];
  const float* Wq    = (const float*)d_in[6];
  const float* bq    = (const float*)d_in[7];
  const float* Wk    = (const float*)d_in[8];
  const float* bk    = (const float*)d_in[9];
  const float* Wv    = (const float*)d_in[10];
  const float* bv    = (const float*)d_in[11];
  const float* ln2_g = (const float*)d_in[12];
  const float* ln2_b = (const float*)d_in[13];
  const float* W1    = (const float*)d_in[14];
  const float* b1    = (const float*)d_in[15];
  const float* W2    = (const float*)d_in[16];
  const float* b2    = (const float*)d_in[17];

  char* w = (char*)d_ws;
  size_t off = 0;
  auto alloc = [&](size_t bytes) -> void* {
    void* p = w + off;
    off = (off + bytes + 255) & ~(size_t)255;
    return p;
  };

  unsigned short* wWp  = (unsigned short*)alloc((size_t)DMODEL * KPATCH * 2);          // [768][16384]
  unsigned short* wWq  = (unsigned short*)alloc((size_t)LAYERS * DMODEL * DMODEL * 2); // per l: [768][768]
  unsigned short* wWk  = (unsigned short*)alloc((size_t)LAYERS * DMODEL * DMODEL * 2);
  unsigned short* wWv  = (unsigned short*)alloc((size_t)LAYERS * DMODEL * DMODEL * 2);
  unsigned short* wW1  = (unsigned short*)alloc((size_t)LAYERS * DMODEL * DFF * 2);    // per l: [3072][768]
  unsigned short* wW2  = (unsigned short*)alloc((size_t)LAYERS * DFF * DMODEL * 2);    // per l: [768][3072]
  unsigned short* Ap   = (unsigned short*)alloc((size_t)BN * KPATCH * 2);              // im2col
  float*          hbuf = (float*)alloc((size_t)BN * DMODEL * 4);
  unsigned short* ybuf = (unsigned short*)alloc((size_t)BN * DMODEL * 2);
  unsigned short* qb   = (unsigned short*)alloc((size_t)BN * DMODEL * 2);
  unsigned short* kb   = (unsigned short*)alloc((size_t)BN * DMODEL * 2);
  unsigned short* vb   = (unsigned short*)alloc((size_t)BN * DMODEL * 2);
  float*          sc   = (float*)alloc((size_t)BATCH * HEADS * NTOK * NTOK * 4);
  unsigned short* pb   = (unsigned short*)alloc((size_t)BATCH * HEADS * NTOK * NTOK * 2);
  unsigned short* hid  = (unsigned short*)alloc((size_t)BN * DFF * 2);

  // ---- one-time conversions ----
  {
    long tot = (long)DMODEL * KPATCH;
    flat_cvt_bf16<<<(tot + 255) / 256, 256, 0, stream>>>(Wp, wWp, tot);
  }
  {
    long tot = (long)LAYERS * HEADS * DMODEL * HDIM;   // batch=L*H, [768][64] -> [64][768]
    transpose_cvt_bf16<<<(tot + 255) / 256, 256, 0, stream>>>(Wq, wWq, DMODEL, HDIM, tot);
    transpose_cvt_bf16<<<(tot + 255) / 256, 256, 0, stream>>>(Wk, wWk, DMODEL, HDIM, tot);
    transpose_cvt_bf16<<<(tot + 255) / 256, 256, 0, stream>>>(Wv, wWv, DMODEL, HDIM, tot);
  }
  {
    long tot = (long)LAYERS * DMODEL * DFF;            // batch=L, [768][3072] -> [3072][768]
    transpose_cvt_bf16<<<(tot + 255) / 256, 256, 0, stream>>>(W1, wW1, DMODEL, DFF, tot);
    transpose_cvt_bf16<<<(tot + 255) / 256, 256, 0, stream>>>(W2, wW2, DFF, DMODEL, tot);
  }
  im2col_bf16<<<(NPIX + 255) / 256, 256, 0, stream>>>(x, Ap);

  // ---- patch embedding: h = Ap * wWp^T + bp ; then + pos ----
  launch_gemm_t<0, 0>(stream, Ap, wWp, hbuf, nullptr, bp,
                      BN, DMODEL, KPATCH, KPATCH, KPATCH, DMODEL,
                      1.0f, 1, 1, 0, 0, 0, 0, 0, 0);
  add_pos_kernel<<<((long)BN * DMODEL + 255) / 256, 256, 0, stream>>>(hbuf, pos);

  // ---- transformer layers ----
  for (int l = 0; l < LAYERS; ++l) {
    // LN1
    layernorm_bf16<<<BN, 256, 0, stream>>>(hbuf, ln1_g + (long)l * DMODEL,
                                           ln1_b + (long)l * DMODEL, ybuf);
    // QKV projections (bf16 out, bias folded)
    launch_gemm_t<0, 2>(stream, ybuf, wWq + (long)l * DMODEL * DMODEL, nullptr, qb,
                        bq + (long)l * DMODEL, BN, DMODEL, DMODEL, DMODEL, DMODEL, DMODEL,
                        1.0f, 1, 1, 0, 0, 0, 0, 0, 0);
    launch_gemm_t<0, 2>(stream, ybuf, wWk + (long)l * DMODEL * DMODEL, nullptr, kb,
                        bk + (long)l * DMODEL, BN, DMODEL, DMODEL, DMODEL, DMODEL, DMODEL,
                        1.0f, 1, 1, 0, 0, 0, 0, 0, 0);
    launch_gemm_t<0, 2>(stream, ybuf, wWv + (long)l * DMODEL * DMODEL, nullptr, vb,
                        bv + (long)l * DMODEL, BN, DMODEL, DMODEL, DMODEL, DMODEL, DMODEL,
                        1.0f, 1, 1, 0, 0, 0, 0, 0, 0);
    // S = q k^T / 8  (48 batched 512x512x64, k consumed directly as B^T)
    launch_gemm_t<0, 0>(stream, qb, kb, sc, nullptr, nullptr,
                        NTOK, NTOK, HDIM, DMODEL, DMODEL, NTOK,
                        0.125f, BATCH * HEADS, HEADS,
                        (long)NTOK * DMODEL, HDIM,
                        (long)NTOK * DMODEL, HDIM,
                        (long)HEADS * NTOK * NTOK, (long)NTOK * NTOK);
    // softmax rows -> bf16 probs
    softmax_bf16<<<BATCH * HEADS * NTOK, 128, 0, stream>>>(sc, pb);
    // O = P V  (row-major B path), residual-added into h with head concat
    launch_gemm_t<1, 1>(stream, pb, vb, hbuf, nullptr, nullptr,
                        NTOK, HDIM, NTOK, NTOK, DMODEL, DMODEL,
                        1.0f, BATCH * HEADS, HEADS,
                        (long)HEADS * NTOK * NTOK, (long)NTOK * NTOK,
                        (long)NTOK * DMODEL, HDIM,
                        (long)NTOK * DMODEL, HDIM);
    // LN2
    layernorm_bf16<<<BN, 256, 0, stream>>>(hbuf, ln2_g + (long)l * DMODEL,
                                           ln2_b + (long)l * DMODEL, ybuf);
    // MLP1: gelu(y W1 + b1) -> bf16 hidden
    launch_gemm_t<0, 3>(stream, ybuf, wW1 + (long)l * DMODEL * DFF, nullptr, hid,
                        b1 + (long)l * DFF, BN, DFF, DMODEL, DMODEL, DMODEL, DFF,
                        1.0f, 1, 1, 0, 0, 0, 0, 0, 0);
    // MLP2: h += hid W2 + b2
    launch_gemm_t<0, 1>(stream, hid, wW2 + (long)l * DFF * DMODEL, hbuf, nullptr,
                        b2 + (long)l * DMODEL, BN, DMODEL, DFF, DFF, DFF, DMODEL,
                        1.0f, 1, 1, 0, 0, 0, 0, 0, 0);
  }

  // ---- output ----
  long tot = (long)BN * DMODEL;
  copy_f32<<<(tot + 255) / 256, 256, 0, stream>>>(hbuf, (float*)d_out, tot);
}
